// MlpwithSOMModule_8710193676348
// MI455X (gfx1250) — matrix operand order repaired and checked by reading, not compiled, not run on hardware
//
#include <hip/hip_runtime.h>

// MI455X / gfx1250: wave32, WMMA f16 16x16x32, f32 accumulate.
// out[pair] = sum_l max_m ( ctx[l,:] . ent[m,:] ),  pair = b*64+k.
// Memory-roofline design: ctx read once (register-resident A stripes, f16),
// ent read twice (LDS-staged B tiles, double-buffered, f16), ~17-26us floor.

typedef _Float16 v16h __attribute__((ext_vector_type(16)));
typedef _Float16 h8   __attribute__((ext_vector_type(8)));
typedef _Float16 h4   __attribute__((ext_vector_type(4)));
typedef _Float16 h2   __attribute__((ext_vector_type(2)));
typedef __fp16   f16x2 __attribute__((ext_vector_type(2)));
typedef float    v8f  __attribute__((ext_vector_type(8)));

#define D        768
#define LROWS    256
#define LDSTRIDE 776                    // 768 + 8 halfs pad -> 4-bank row skew
#define TILEH    (16 * LDSTRIDE)        // one 16-row ent tile in halfs

static __device__ inline h2 pkh2(float lo, float hi) {
  f16x2 t = __builtin_amdgcn_cvt_pkrtz(lo, hi);   // v_cvt_pk_rtz_f16_f32
  return __builtin_bit_cast(h2, t);
}

static __device__ inline h8 cvt8(float4 f0, float4 f1) {
  h2 a = pkh2(f0.x, f0.y);
  h2 b = pkh2(f0.z, f0.w);
  h2 c = pkh2(f1.x, f1.y);
  h2 d = pkh2(f1.z, f1.w);
  h4 ab = __builtin_shufflevector(a, b, 0, 1, 2, 3);
  h4 cd = __builtin_shufflevector(c, d, 0, 1, 2, 3);
  return __builtin_shufflevector(ab, cd, 0, 1, 2, 3, 4, 5, 6, 7);
}

__global__ __launch_bounds__(256, 1)
void som_rowmax_gemm(const float* __restrict__ context, float* __restrict__ out)
{
  __shared__ __align__(16) _Float16 entLDS[2 * TILEH];   // double-buffered n-tile
  __shared__ float partial[8];

  const int pair = blockIdx.x;                       // 0..511  (b*64+k)
  const float* ctxBase = context + (size_t)pair * 2u * LROWS * D;
  const float* entBase = ctxBase + (size_t)LROWS * D;

  const int tid  = threadIdx.x;
  const int wave = tid >> 5;
  const int lane = tid & 31;
  const int hi   = lane >> 4;                        // half-wave select
  const int l16  = lane & 15;
  const int sr   = tid >> 4;                         // staging row 0..15
  const int sc   = tid & 15;                         // staging col group

  float total = 0.0f;

  for (int pass = 0; pass < 2; ++pass) {
    // ---- A stripe: this wave's 16 ctx rows, full K=768, f16 in registers.
    // A layout (16x32 f16): lane<16 holds row M=l16, K {d0..d0+7, d0+16..d0+23};
    // lane>=16 same rows, K shifted +8. All chunks contiguous in d.
    const int rowA = pass * 128 + wave * 16 + l16;
    const float* aRow = ctxBase + (size_t)rowA * D;

    v16h Areg[24];
    #pragma unroll
    for (int g = 0; g < 6; ++g) {                    // 16 b128 loads in flight/group
      float4 fb[16];
      #pragma unroll
      for (int u = 0; u < 4; ++u) {
        const int d0 = 32 * (4 * g + u) + 8 * hi;
        fb[4 * u + 0] = *(const float4*)(aRow + d0);
        fb[4 * u + 1] = *(const float4*)(aRow + d0 + 4);
        fb[4 * u + 2] = *(const float4*)(aRow + d0 + 16);
        fb[4 * u + 3] = *(const float4*)(aRow + d0 + 20);
      }
      #pragma unroll
      for (int u = 0; u < 4; ++u) {
        h8 alo = cvt8(fb[4 * u + 0], fb[4 * u + 1]);
        h8 ahi = cvt8(fb[4 * u + 2], fb[4 * u + 3]);
        Areg[4 * g + u] = __builtin_shufflevector(alo, ahi, 0, 1, 2, 3, 4, 5, 6, 7,
                                                  8, 9, 10, 11, 12, 13, 14, 15);
      }
    }

    // ---- Prologue: stage ent tile 0 into LDS buffer 0 (grouped loads).
    {
      const float* t0 = entBase + (size_t)sr * D;
      #pragma unroll
      for (int ch = 0; ch < 3; ++ch) {
        float4 st[4];
        #pragma unroll
        for (int u = 0; u < 4; ++u)
          st[u] = *(const float4*)(t0 + ((sc + 16 * (ch * 4 + u)) << 2));
        #pragma unroll
        for (int u = 0; u < 4; ++u) {
          const int cf = (sc + 16 * (ch * 4 + u)) << 2;
          h2 p0 = pkh2(st[u].x, st[u].y);
          h2 p1 = pkh2(st[u].z, st[u].w);
          *(h4*)(&entLDS[sr * LDSTRIDE + cf]) =
              __builtin_shufflevector(p0, p1, 0, 1, 2, 3);
        }
      }
    }
    __syncthreads();

    v8f rmax;
    #pragma unroll
    for (int v = 0; v < 8; ++v) rmax[v] = -3.402823466e38f;

    for (int j = 0; j < 16; ++j) {                   // n-tiles: m = 16j..16j+15
      const bool stg = (j < 15);
      // B layout (32x16 f16): lane n = column m=16j+n: K {0..7,16..23} (lanes<16) or
      // K {8..15,24..31} (lanes>=16), matching the A layout.
      const _Float16* bRow = &entLDS[(j & 1) * TILEH + l16 * LDSTRIDE + 8 * hi];
      _Float16* nDst = &entLDS[((j + 1) & 1) * TILEH + sr * LDSTRIDE];
      const float* nTile = entBase + (size_t)(((j + 1) & 15) * 16) * D + (size_t)sr * D;

      v8f acc = {0.f, 0.f, 0.f, 0.f, 0.f, 0.f, 0.f, 0.f};
      // B register pipeline, issue distance 2 (3 live h8 pairs).
      h8 blo[24], bhi[24];
      blo[0] = *(const h8*)(bRow);
      bhi[0] = *(const h8*)(bRow + 16);
      blo[1] = *(const h8*)(bRow + 32);
      bhi[1] = *(const h8*)(bRow + 48);
      float4 st[4];

      #pragma unroll
      for (int t = 0; t < 24; ++t) {
        if (t + 2 < 24) {
          blo[t + 2] = *(const h8*)(bRow + 32 * (t + 2));       // ds_load_b128
          bhi[t + 2] = *(const h8*)(bRow + 32 * (t + 2) + 16);  // ds_load_b128
        }
        // Pipelined staging of next ent tile: issue 4 global loads at t=0/8/16,
        // convert+store them at t=7/15/23 (latency hidden behind 8 WMMAs).
        if ((t & 7) == 0 && stg) {
          const int cb = (t >> 3) * 4;
          #pragma unroll
          for (int u = 0; u < 4; ++u)
            st[u] = *(const float4*)(nTile + ((sc + 16 * (cb + u)) << 2));
        }
        v16h bb = __builtin_shufflevector(blo[t], bhi[t], 0, 1, 2, 3, 4, 5, 6, 7,
                                          8, 9, 10, 11, 12, 13, 14, 15);
        acc = __builtin_amdgcn_wmma_f32_16x16x32_f16(
                  false, Areg[t], false, bb, (short)0, acc, false, false);
        if ((t & 7) == 7 && stg) {
          const int cb = (t >> 3) * 4;
          #pragma unroll
          for (int u = 0; u < 4; ++u) {
            const int cf = (sc + 16 * (cb + u)) << 2;
            h2 p0 = pkh2(st[u].x, st[u].y);
            h2 p1 = pkh2(st[u].z, st[u].w);
            *(h4*)(nDst + cf) = __builtin_shufflevector(p0, p1, 0, 1, 2, 3);
          }
        }
      }

      // Elementwise max across n-tiles keeps per-(row, n mod 16) partial maxima
      // directly in the C layout (VGPR v, half h) <-> row v+8h.
      #pragma unroll
      for (int v = 0; v < 8; ++v) rmax[v] = fmaxf(rmax[v], acc[v]);

      __syncthreads();   // next-tile writes complete; prior-buffer reads consumed
    }

    // Cross-lane max over the 16 lanes of each half -> true row maxima; sum rows.
    float s = 0.0f;
    #pragma unroll
    for (int v = 0; v < 8; ++v) {
      float x = rmax[v];
      x = fmaxf(x, __shfl_xor(x, 1, 32));
      x = fmaxf(x, __shfl_xor(x, 2, 32));
      x = fmaxf(x, __shfl_xor(x, 4, 32));
      x = fmaxf(x, __shfl_xor(x, 8, 32));
      s += x;                                        // lanes<16: rows v; >=16: rows v+8
    }
    s += __shfl_xor(s, 16, 32);                      // all lanes: sum of 16 rows
    total += s;
  }

  if (lane == 0) partial[wave] = total;
  __syncthreads();
  if (tid == 0) {
    float acc_out = 0.0f;
    #pragma unroll
    for (int w = 0; w < 8; ++w) acc_out += partial[w];
    out[pair] = acc_out;
  }
}

extern "C" void kernel_launch(void* const* d_in, const int* in_sizes, int n_in,
                              void* d_out, int out_size, void* d_ws, size_t ws_size,
                              hipStream_t stream) {
  (void)in_sizes; (void)n_in; (void)d_ws; (void)ws_size; (void)out_size;
  const float* context = (const float*)d_in[0];
  float* out = (float*)d_out;
  som_rowmax_gemm<<<dim3(512), dim3(256), 0, stream>>>(context, out);
}
